// DeepGRU_52080773431930
// MI455X (gfx1250) — matrix operand order repaired and checked by reading, not compiled, not run on hardware
//
#include <hip/hip_runtime.h>
#include <hip/hip_bf16.h>

#define BATCHN 256
#define SEQN   2048
#define HID    128
#define NLAYER 5
#define WPERL  (6 * 4 * 8 * 32 * 16)   // 98304 packed bf16 elements per layer

typedef __attribute__((ext_vector_type(16))) __bf16 v16bf;
typedef __attribute__((ext_vector_type(8)))  float  v8f;

__device__ __forceinline__ unsigned short f2bf(float x) {
  unsigned u = __builtin_bit_cast(unsigned, x);
  u += 0x7FFFu + ((u >> 16) & 1u);           // round-to-nearest-even
  return (unsigned short)(u >> 16);
}
__device__ __forceinline__ float bf2f(unsigned short b) {
  unsigned u = ((unsigned)b) << 16;
  return __builtin_bit_cast(float, u);
}

union Frag32B { uint4 q[2]; v16bf v; };

// A-fragment (16x32 bf16, MxK): two aligned 16B LDS loads off a per-lane base.
__device__ __forceinline__ v16bf load_a(const unsigned short* lane_base, int elem_off) {
  Frag32B f;
  f.q[0] = *(const uint4*)(lane_base + elem_off);        // K = kk*32 + kb .. +7
  f.q[1] = *(const uint4*)(lane_base + elem_off + 16);   // K = kk*32 + 16 + kb .. +7
  return f.v;
}

// B-fragment (32x16 bf16, KxN): 32 contiguous bytes per lane, one base +
// compile-time element offset (lowers to global_load_b128 ... offset:imm).
__device__ __forceinline__ v16bf load_b(const unsigned short* lane_base, int elem_off) {
  Frag32B f;
  f.q[0] = *(const uint4*)(lane_base + elem_off);
  f.q[1] = *(const uint4*)(lane_base + elem_off + 8);
  return f.v;
}

// ---------------- weight pre-pack: f32 [L][128][128] -> bf16 WMMA-B fragments ----------------
__global__ __launch_bounds__(256) void pack_weights(
    const float* __restrict__ Whr, const float* __restrict__ Wxr,
    const float* __restrict__ Whz, const float* __restrict__ Wxz,
    const float* __restrict__ Whh, const float* __restrict__ Wxh,
    unsigned short* __restrict__ packed) {
  int idx = blockIdx.x * 256 + threadIdx.x;              // < 5*98304 = 491520
  int l    = idx / WPERL;
  int r    = idx % WPERL;
  int mat  = r / 16384;
  int r2   = r % 16384;
  int kk   = r2 >> 12;
  int nt   = (r2 >> 9) & 7;
  int lane = (r2 >> 4) & 31;
  int j    = r2 & 15;
  int K = kk * 32 + ((lane & 16) ? 8 : 0) + (j < 8 ? j : 8 + j);
  int N = nt * 16 + (lane & 15);
  const float* W;
  switch (mat) {
    case 0: W = Whr; break;
    case 1: W = Wxr; break;
    case 2: W = Whz; break;
    case 3: W = Wxz; break;
    case 4: W = Whh; break;
    default: W = Wxh; break;
  }
  packed[idx] = f2bf(W[((size_t)l * HID + K) * HID + N]);
}

// ---------------- main persistent GRU kernel: 1 block = 16 batch rows, 8 waves = 8 N-tiles ----
__global__ __launch_bounds__(256) void gru_main(
    const float* __restrict__ x_in,
    const float* __restrict__ b_r, const float* __restrict__ b_z, const float* __restrict__ b_h,
    const unsigned short* __restrict__ wpack,
    float* __restrict__ out) {
  __shared__ __align__(16) unsigned short h_s[NLAYER][16][HID];  // persistent h per layer (bf16)
  __shared__ __align__(16) unsigned short xin_s[16][HID];        // current layer input
  __shared__ __align__(16) unsigned short rh_s[16][HID];         // R * h_prev
  __shared__ float bias_s[3][NLAYER * HID];                      // r/z/h biases (t-invariant)

  const int tid  = threadIdx.x;
  const int lane = tid & 31;
  const int wv   = tid >> 5;                             // wave id = N-tile id (0..7)
  const int wvs  = __builtin_amdgcn_readfirstlane(wv);   // provably wave-uniform

  for (int i = tid; i < NLAYER * 16 * HID; i += 256)
    ((unsigned short*)h_s)[i] = 0;
  for (int i = tid; i < NLAYER * HID; i += 256) {
    bias_s[0][i] = b_r[i];
    bias_s[1][i] = b_z[i];
    bias_s[2][i] = b_h[i];
  }
  __syncthreads();

  const int bt = blockIdx.x;                 // batch tile (16 rows)
  const int N  = wvs * 16 + (lane & 15);     // C-fragment: lane -> column
  const int Mb = (lane & 16) ? 8 : 0;        // C-fragment: vgpr v -> row Mb+v

  // Per-lane bases computed ONCE.
  const unsigned short* wlane0 = wpack + (size_t)(wvs * 512) + lane * 16;
  const int aoff = (lane & 15) * HID + ((lane & 16) ? 8 : 0);
  const unsigned short* xa = &xin_s[0][0] + aoff;
  const unsigned short* ra = &rh_s[0][0] + aoff;

#pragma unroll 1
  for (int t = 0; t < SEQN; ++t) {
    // stage x_t (f32 -> bf16) into LDS; streamed once -> non-temporal
    for (int i = tid; i < 16 * HID; i += 256) {
      int r = i >> 7, c = i & 127;
      float xv = __builtin_nontemporal_load(
          &x_in[((size_t)(bt * 16 + r) * SEQN + t) * HID + c]);
      xin_s[r][c] = f2bf(xv);
    }
    // prefetch next timestep's x slab (16 rows x 512B); one inst covers it lane-wise
    if (t + 1 < SEQN) {
      const float* nx = x_in + ((size_t)(bt * 16 + (tid >> 4)) * SEQN + (t + 1)) * HID
                             + (tid & 15) * 8;
      __builtin_prefetch(nx, 0, 3);
    }
    __syncthreads();

    const unsigned short* wl = wlane0;                 // += WPERL per layer
    const unsigned short* ha = &h_s[0][0][0] + aoff;   // += 2048 per layer

#pragma unroll 1
    for (int l = 0; l < NLAYER; ++l) {
      v8f accR = {}; v8f accZ = {}; v8f accC = {};

      // phase 1: R, Z pre-activations + xin @ W_xh (interleaved accumulators)
#pragma unroll
      for (int kk = 0; kk < 4; ++kk) {
        v16bf a_h  = load_a(ha, kk * 32);
        v16bf a_x  = load_a(xa, kk * 32);
        v16bf b_hr = load_b(wl, 0 * 16384 + kk * 4096);
        v16bf b_xr = load_b(wl, 1 * 16384 + kk * 4096);
        v16bf b_hz = load_b(wl, 2 * 16384 + kk * 4096);
        v16bf b_xz = load_b(wl, 3 * 16384 + kk * 4096);
        v16bf b_xh = load_b(wl, 5 * 16384 + kk * 4096);
        accR = __builtin_amdgcn_wmma_f32_16x16x32_bf16(false, a_h, false, b_hr, (short)0, accR, false, false);
        accZ = __builtin_amdgcn_wmma_f32_16x16x32_bf16(false, a_h, false, b_hz, (short)0, accZ, false, false);
        accC = __builtin_amdgcn_wmma_f32_16x16x32_bf16(false, a_x, false, b_xh, (short)0, accC, false, false);
        accR = __builtin_amdgcn_wmma_f32_16x16x32_bf16(false, a_x, false, b_xr, (short)0, accR, false, false);
        accZ = __builtin_amdgcn_wmma_f32_16x16x32_bf16(false, a_x, false, b_xz, (short)0, accZ, false, false);
      }

      // pull the NEXT layer's weight chunks toward this WGP while we do gate math +
      // barrier (prefetch has no counter / no dest, so it is off the critical path)
      if (l < NLAYER - 1) {
        const unsigned short* wn = wl + WPERL;
#pragma unroll
        for (int mat = 0; mat < 6; ++mat)
#pragma unroll
          for (int kk = 0; kk < 4; ++kk)
            __builtin_prefetch(wn + mat * 16384 + kk * 4096, 0, 3);
      }

      const float brv = bias_s[0][l * HID + N];
      const float bzv = bias_s[1][l * HID + N];
      const float bhv = bias_s[2][l * HID + N];

      float R[8], Z[8], HP[8];
#pragma unroll
      for (int v = 0; v < 8; ++v) {
        HP[v] = bf2f(h_s[l][Mb + v][N]);
        R[v]  = 1.0f / (1.0f + __expf(-(accR[v] + brv)));
        Z[v]  = 1.0f / (1.0f + __expf(-(accZ[v] + bzv)));
        rh_s[Mb + v][N] = f2bf(R[v] * HP[v]);
      }
      __syncthreads();   // rh_s complete; all phase-1 reads of h_s/xin_s done

      // phase 2: (R*h) @ W_hh — two accumulators to halve the WMMA dep chain
      v8f accD = {};
#pragma unroll
      for (int kk = 0; kk < 4; ++kk) {
        v16bf a_rh = load_a(ra, kk * 32);
        v16bf b_hh = load_b(wl, 4 * 16384 + kk * 4096);
        if (kk & 1)
          accD = __builtin_amdgcn_wmma_f32_16x16x32_bf16(false, a_rh, false, b_hh, (short)0, accD, false, false);
        else
          accC = __builtin_amdgcn_wmma_f32_16x16x32_bf16(false, a_rh, false, b_hh, (short)0, accC, false, false);
      }

#pragma unroll
      for (int v = 0; v < 8; ++v) {
        float ct = tanhf(accC[v] + accD[v] + bhv);
        float ht = Z[v] * ct + (1.0f - Z[v]) * HP[v];
        unsigned short hb = f2bf(ht);
        h_s[l][Mb + v][N] = hb;   // state for t+1
        xin_s[Mb + v][N]  = hb;   // input for layer l+1
        if (l == NLAYER - 1)
          __builtin_nontemporal_store(
              ht, &out[((size_t)(bt * 16 + Mb + v) * SEQN + t) * HID + N]);
      }
      __syncthreads();   // xin_s/h_s ready for next layer / next step

      wl += WPERL;
      ha += 16 * HID;
    }
  }
}

extern "C" void kernel_launch(void* const* d_in, const int* in_sizes, int n_in,
                              void* d_out, int out_size, void* d_ws, size_t ws_size,
                              hipStream_t stream) {
  (void)in_sizes; (void)n_in; (void)out_size; (void)ws_size;
  const float* x   = (const float*)d_in[0];
  const float* Whr = (const float*)d_in[1];
  const float* Wxr = (const float*)d_in[2];
  const float* br  = (const float*)d_in[3];
  const float* Whz = (const float*)d_in[4];
  const float* Wxz = (const float*)d_in[5];
  const float* bz  = (const float*)d_in[6];
  const float* Whh = (const float*)d_in[7];
  const float* Wxh = (const float*)d_in[8];
  const float* bh  = (const float*)d_in[9];
  unsigned short* wpack = (unsigned short*)d_ws;   // 491520 * 2B < 1 MB
  float* out = (float*)d_out;

  const int total = NLAYER * WPERL;                // 491520
  pack_weights<<<total / 256, 256, 0, stream>>>(Whr, Wxr, Whz, Wxz, Whh, Wxh, wpack);
  gru_main<<<BATCHN / 16, 256, 0, stream>>>(x, br, bz, bh, wpack, out);
}
